// GatedInceptionBlock_7146825580711
// MI455X (gfx1250) — matrix operand order repaired and checked by reading, not compiled, not run on hardware
//
#include <hip/hip_runtime.h>
#include <hip/hip_bf16.h>
#include <math.h>

// ---------- types for WMMA ----------
typedef __attribute__((ext_vector_type(16))) __bf16          v16bf;
typedef __attribute__((ext_vector_type(8)))  float           v8f;
typedef __attribute__((ext_vector_type(16))) unsigned short  u16x16;
typedef __attribute__((ext_vector_type(8)))  unsigned short  u16x8;

union BFV { u16x16 u; v16bf b; unsigned short s[16]; };

__device__ __forceinline__ unsigned short f2bf(float f) {
  union { float f; unsigned int u; } c; c.f = f;
  unsigned int x = c.u;
  return (unsigned short)((x + 0x7FFFu + ((x >> 16) & 1u)) >> 16);
}
__device__ __forceinline__ float bf2f(unsigned short h) {
  union { float f; unsigned int u; } c; c.u = ((unsigned int)h) << 16; return c.f;
}

// ---------- elementwise helpers ----------
__global__ void k_f2bf(const float* __restrict__ src, unsigned short* __restrict__ dst, int n) {
  int i = blockIdx.x * blockDim.x + threadIdx.x;
  if (i < n) dst[i] = f2bf(src[i]);
}
__global__ void k_copy_f32(const float* __restrict__ src, float* __restrict__ dst, int n) {
  int i = blockIdx.x * blockDim.x + threadIdx.x;
  if (i < n) dst[i] = src[i];
}
__global__ void k_zero_f32(float* __restrict__ p, int n) {
  int i = blockIdx.x * blockDim.x + threadIdx.x;
  if (i < n) p[i] = 0.0f;
}

// ---------- weight packer: W[K][64][Cout] fp32 -> WMMA B-operand bf16 layout ----------
__global__ void k_pack(const float* __restrict__ W, unsigned short* __restrict__ dst,
                       int NCtot, int total) {
  int e = blockIdx.x * blockDim.x + threadIdx.x;
  if (e >= total) return;
  int j    = e & 15;
  int lane = (e >> 4) & 31;
  int nct  = (e >> 9) % NCtot;
  int rest = (e >> 9) / NCtot;
  int cch  = rest & 1;
  int k    = rest >> 1;
  int c    = cch * 32 + (lane >> 4) * 16 + j;   // input channel
  int o    = nct * 16 + (lane & 15);            // output channel
  int Cout = NCtot * 16;
  dst[e] = f2bf(W[((size_t)k * 64 + c) * Cout + o]);
}

// ---------- WMMA GEMM: out[N,Cout] = gather(X,nbr)[N,K*64] x Wpacked ----------
// 4 waves / block, M-tile = 64 rows (16 per wave), B panel double-buffered in LDS.
// Software pipeline per k-step:
//   global loads of panel k+1 -> regs S   (latency hidden by WMMAs)
//   ds_load all 8 B tiles of panel k      (one small dscnt wait)
//   global loads of A rows for k+1        (consumed next step)
//   8 back-to-back WMMAs
//   ds_store S -> other LDS buffer, barrier
template <int HAS_NBR>
__global__ void k_wmma_gemm(const unsigned short* __restrict__ X,   // [N,64] bf16
                            const int* __restrict__ nbr, int K,     // [N,K]
                            const unsigned short* __restrict__ Bp, int NCtot,
                            const float* __restrict__ scale,
                            const float* __restrict__ bias,
                            int Cout,
                            float* __restrict__ outF,
                            unsigned short* __restrict__ outB,
                            int relu_flag, int accum_flag) {
  __shared__ unsigned short ldsB[2][4096];       // 2 x 8KB panels
  const int tid  = threadIdx.x;                  // 0..127
  const int lane = tid & 31;
  const int wave = tid >> 5;                     // 0..3
  const int m    = lane & 15;
  const int hi   = lane >> 4;
  const int n0   = blockIdx.x * 64 + wave * 16;  // this wave's 16-row tile
  const int ncb  = blockIdx.y << 2;              // first 16-col tile

  const v8f vzero = {0.f,0.f,0.f,0.f,0.f,0.f,0.f,0.f};
  v8f acc[4] = {vzero, vzero, vzero, vzero};

  u16x8 S[4];                                    // staged panel chunks (regs)
  auto loadS = [&](int k) {
#pragma unroll
    for (int it = 0; it < 4; ++it) {
      int c16 = tid + it * 128;                  // 0..511 16-byte chunks
      int seg = c16 >> 8, off = c16 & 255;
      S[it] = *(const u16x8*)(Bp +
          ((size_t)(k * 2 + seg) * NCtot + ncb) * 512 + (size_t)off * 8);
    }
  };
  auto storeS = [&](int buf) {
#pragma unroll
    for (int it = 0; it < 4; ++it) {
      int c16 = tid + it * 128;
      int seg = c16 >> 8, off = c16 & 255;
      *(u16x8*)(&ldsB[buf][seg * 2048 + off * 8]) = S[it];
    }
  };
  auto loadA = [&](BFV* Ar, const unsigned short* xr) {
#pragma unroll
    for (int cch = 0; cch < 2; ++cch) {
      const unsigned short* a0 = xr + cch * 32 + hi * 16;
      u16x8 lo = *(const u16x8*)(a0);
      u16x8 hh = *(const u16x8*)(a0 + 8);
#pragma unroll
      for (int j = 0; j < 8; ++j) { Ar[cch].s[j] = lo[j]; Ar[cch].s[8 + j] = hh[j]; }
    }
  };

  const size_t nbase = (size_t)(n0 + m) * (size_t)K;
  auto rowIdx = [&](int k) -> int {
    if (!HAS_NBR) return n0 + m;
    int kk = (k < K) ? k : (K - 1);              // branchless clamp
    return nbr[nbase + kk];
  };

  // prologue: panel 0 into LDS, A row for k=0, row index for k=1
  loadS(0); storeS(0);
  BFV A[2];
  loadA(A, X + (size_t)rowIdx(0) * 64);
  int row_nxt = rowIdx(1);
  __syncthreads();

  for (int k = 0; k < K; ++k) {
    const int cur = k & 1;
    const bool more = (k + 1 < K);
    if (more) loadS(k + 1);                      // global loads, consumed post-WMMA

    BFV B[8];                                    // all 8 B tiles up front
#pragma unroll
    for (int t = 0; t < 8; ++t)
      B[t].u = *(const u16x16*)(&ldsB[cur][(t >> 2) * 2048 + (t & 3) * 512 + lane * 16]);

    BFV An[2];
    if (more) loadA(An, X + (size_t)row_nxt * 64);  // A for k+1
    int row_n2 = rowIdx(k + 2);                     // index for k+2
    if (k + 2 < K)                                  // pure-arithmetic prefetch of panel k+2
      __builtin_prefetch(Bp + ((size_t)(k + 2) * 2 * NCtot + ncb) * 512 +
                         (size_t)lane * 128, 0, 3);

#pragma unroll
    for (int t = 0; t < 8; ++t)
      acc[t & 3] = __builtin_amdgcn_wmma_f32_16x16x32_bf16(
          false, A[t >> 2].b, false, B[t].b, (short)0, acc[t & 3], false, false);

    if (more) {
      storeS(cur ^ 1);                           // commit panel k+1 to LDS
#pragma unroll
      for (int c2 = 0; c2 < 2; ++c2) A[c2] = An[c2];
      row_nxt = row_n2;
    }
    __syncthreads();
  }

  // epilogue: C/D layout -> element (vgpr r, lane): M = r + 8*hi, N = lane&15
#pragma unroll
  for (int nc = 0; nc < 4; ++nc) {
    const int col = (ncb + nc) * 16 + (lane & 15);
    const float sc = scale ? scale[col] : 1.0f;
    const float bs = bias  ? bias[col]  : 0.0f;
#pragma unroll
    for (int r = 0; r < 8; ++r) {
      const int mm = n0 + r + hi * 8;
      float y = acc[nc][r] * sc + bs;
      if (relu_flag) y = fmaxf(y, 0.0f);
      const size_t o = (size_t)mm * Cout + col;
      if (outF) { if (accum_flag) outF[o] += y; else outF[o] = y; }
      if (outB) outB[o] = f2bf(y);
    }
  }
}

// ---------- ECA ----------
__global__ void k_counts(const int* __restrict__ bidx, float* __restrict__ counts, int N) {
  int n = blockIdx.x * blockDim.x + threadIdx.x;
  if (n < N) atomicAdd(&counts[bidx[n]], 1.0f);
}
__global__ void k_sums(const float* __restrict__ x, const int* __restrict__ bidx,
                       float* __restrict__ sums, int n64) {
  int e = blockIdx.x * blockDim.x + threadIdx.x;
  if (e >= n64) return;
  int n = e >> 6, c = e & 63;
  atomicAdd(&sums[bidx[n] * 64 + c], x[e]);
}
__global__ void k_gate(const float* __restrict__ sums, const float* __restrict__ counts,
                       const float* __restrict__ kern, float* __restrict__ gate) {
  int t = blockIdx.x * blockDim.x + threadIdx.x;
  if (t >= 4 * 64) return;
  int b = t >> 6, c = t & 63;
  float cnt = counts[b];
  float pm = (c > 0)  ? sums[b * 64 + c - 1] / cnt : 0.0f;
  float pc = sums[b * 64 + c] / cnt;
  float pp = (c < 63) ? sums[b * 64 + c + 1] / cnt : 0.0f;
  float y = pm * kern[0] + pc * kern[1] + pp * kern[2];
  gate[t] = 1.0f / (1.0f + expf(-y));
}
__global__ void k_apply(float* __restrict__ x, unsigned short* __restrict__ xb,
                        const float* __restrict__ gate, const int* __restrict__ bidx, int n64) {
  int e = blockIdx.x * blockDim.x + threadIdx.x;
  if (e >= n64) return;
  int n = e >> 6, c = e & 63;
  float v = x[e] * gate[bidx[n] * 64 + c];
  x[e] = v;
  xb[e] = f2bf(v);
}

// ---------- final: softmax over 5 branches, top-k mask, gated combine ----------
__global__ void k_final(const float* __restrict__ w320,
                        const float* __restrict__ x0, const float* __restrict__ x1,
                        const float* __restrict__ x2, const float* __restrict__ x3,
                        const float* __restrict__ x4,
                        const int* __restrict__ keep_dim,
                        float* __restrict__ out, int n64) {
  int e = blockIdx.x * blockDim.x + threadIdx.x;
  if (e >= n64) return;
  int n = e >> 6, c = e & 63;
  float f[5];
#pragma unroll
  for (int i = 0; i < 5; ++i) f[i] = w320[(size_t)n * 320 + i * 64 + c];
  float mx = f[0];
#pragma unroll
  for (int i = 1; i < 5; ++i) mx = fmaxf(mx, f[i]);
  float p[5], s = 0.0f;
#pragma unroll
  for (int i = 0; i < 5; ++i) { p[i] = expf(f[i] - mx); s += p[i]; }
#pragma unroll
  for (int i = 0; i < 5; ++i) p[i] /= s;
  int kd = keep_dim[0];
  if (kd < 1) kd = 1; if (kd > 5) kd = 5;
  if (kd < 5) {
    float t[5];
#pragma unroll
    for (int i = 0; i < 5; ++i) t[i] = p[i];
    for (int a = 0; a < 4; ++a)
      for (int b = 0; b < 4 - a; ++b)
        if (t[b] < t[b + 1]) { float tm = t[b]; t[b] = t[b + 1]; t[b + 1] = tm; }
    float thr = t[kd - 1];
#pragma unroll
    for (int i = 0; i < 5; ++i) if (p[i] < thr) p[i] = 0.0f;
  }
  float acc = p[0] * x0[e] + p[1] * x1[e] + p[2] * x2[e] + p[3] * x3[e] + p[4] * x4[e];
  out[e] = acc;
}

// ---------- host orchestration ----------
extern "C" void kernel_launch(void* const* d_in, const int* in_sizes, int n_in,
                              void* d_out, int out_size, void* d_ws, size_t ws_size,
                              hipStream_t stream) {
  const float* feats = (const float*)d_in[0];
  const int*   nbr3  = (const int*)d_in[1];
  const int*   nbr5  = (const int*)d_in[2];
  const int*   bidx  = (const int*)d_in[3];
  const int*   keep  = (const int*)d_in[4];
  const int N   = in_sizes[0] / 64;
  const int K3  = in_sizes[1] / N;   // 27
  const int K5  = in_sizes[2] / N;   // 125
  const int n64 = N * 64;

  auto F = [&](int i) { return (const float*)d_in[i]; };

  // ---- carve workspace ----
  char* wp = (char*)d_ws;
  auto carve = [&](size_t bytes) -> char* {
    char* p = wp; wp += (bytes + 255) & ~(size_t)255; return p;
  };
  unsigned short* featsb = (unsigned short*)carve((size_t)n64 * 2);
  unsigned short* T1     = (unsigned short*)carve((size_t)n64 * 2);
  unsigned short* T2     = (unsigned short*)carve((size_t)n64 * 2);
  const size_t pk_dense = 4096, pk_k3 = (size_t)K3 * 4096, pk_k5 = (size_t)K5 * 4096;
  unsigned short* Pc30 = (unsigned short*)carve(pk_dense * 2);
  unsigned short* Pc31 = (unsigned short*)carve(pk_k3 * 2);
  unsigned short* Pc50 = (unsigned short*)carve(pk_k3 * 2);
  unsigned short* Pc51 = (unsigned short*)carve(pk_k3 * 2);
  unsigned short* Pc70 = (unsigned short*)carve(pk_dense * 2);
  unsigned short* Pc71 = (unsigned short*)carve(pk_k5 * 2);
  unsigned short* Pc72 = (unsigned short*)carve(pk_k3 * 2);
  unsigned short* Pc90 = (unsigned short*)carve(pk_dense * 2);
  unsigned short* Pc91 = (unsigned short*)carve(pk_k5 * 2);
  unsigned short* Pc92 = (unsigned short*)carve(pk_k5 * 2);
  unsigned short* Ptr[10];
  for (int i = 0; i < 10; ++i) Ptr[i] = (unsigned short*)carve(pk_dense * 2);
  unsigned short* Ptr2 = (unsigned short*)carve((size_t)64 * 320 * 2);
  float* X[5]; unsigned short* XB[5];
  for (int i = 0; i < 5; ++i) X[i]  = (float*)carve((size_t)n64 * 4);
  for (int i = 0; i < 5; ++i) XB[i] = (unsigned short*)carve((size_t)n64 * 2);
  float* wsum = (float*)carve((size_t)n64 * 4);
  float* w320 = (float*)carve((size_t)N * 320 * 4);
  float* sums   = (float*)carve(5 * 256 * 4);
  float* counts = (float*)carve(4 * 4);
  float* gates  = (float*)carve(5 * 256 * 4);

  const int TB = 256;
  auto blks = [&](int n) { return (n + TB - 1) / TB; };

  auto pack = [&](const float* W, unsigned short* dst, int K, int Cout) {
    int NCtot = Cout / 16;
    int total = K * 2 * NCtot * 512;
    k_pack<<<blks(total), TB, 0, stream>>>(W, dst, NCtot, total);
  };
  auto gemm = [&](const unsigned short* Xi, const int* nbr, int K,
                  const unsigned short* Bp, const float* sc, const float* bs,
                  int Cout, float* outF, unsigned short* outB, int relu, int accum) {
    dim3 g(N / 64, Cout / 64);
    if (nbr)
      k_wmma_gemm<1><<<g, 128, 0, stream>>>(Xi, nbr, K, Bp, Cout / 16, sc, bs, Cout,
                                            outF, outB, relu, accum);
    else
      k_wmma_gemm<0><<<g, 128, 0, stream>>>(Xi, nullptr, 1, Bp, Cout / 16, sc, bs, Cout,
                                            outF, outB, relu, accum);
  };

  // ---- prep: bf16 feats + packed bf16 weights ----
  k_f2bf<<<blks(n64), TB, 0, stream>>>(feats, featsb, n64);
  pack(F(7),  Pc30, 1,  64);  pack(F(10), Pc31, K3, 64);
  pack(F(13), Pc50, K3, 64);  pack(F(16), Pc51, K3, 64);
  pack(F(19), Pc70, 1,  64);  pack(F(22), Pc71, K5, 64);  pack(F(25), Pc72, K3, 64);
  pack(F(28), Pc90, 1,  64);  pack(F(31), Pc91, K5, 64);  pack(F(34), Pc92, K5, 64);
  for (int i = 0; i < 5; ++i) {
    pack(F(40 + i * 6 + 2), Ptr[i * 2],     1, 64);
    pack(F(40 + i * 6 + 5), Ptr[i * 2 + 1], 1, 64);
  }
  pack(F(72), Ptr2, 1, 320);

  // ---- five inception branches ----
  gemm(featsb, nullptr, 1,  Pc30, F(6),  F(5),  64, nullptr, T1, 1, 0);
  gemm(T1,     nbr3,    K3, Pc31, F(9),  F(8),  64, X[1], nullptr, 0, 0);
  gemm(featsb, nbr3,    K3, Pc50, F(12), F(11), 64, nullptr, T1, 1, 0);
  gemm(T1,     nbr3,    K3, Pc51, F(15), F(14), 64, X[2], nullptr, 0, 0);
  gemm(featsb, nullptr, 1,  Pc70, F(18), F(17), 64, nullptr, T1, 1, 0);
  gemm(T1,     nbr5,    K5, Pc71, F(21), F(20), 64, nullptr, T2, 1, 0);
  gemm(T2,     nbr3,    K3, Pc72, F(24), F(23), 64, X[3], nullptr, 0, 0);
  gemm(featsb, nullptr, 1,  Pc90, F(27), F(26), 64, nullptr, T1, 1, 0);
  gemm(T1,     nbr5,    K5, Pc91, F(30), F(29), 64, nullptr, T2, 1, 0);
  gemm(T2,     nbr5,    K5, Pc92, F(33), F(32), 64, X[4], nullptr, 0, 0);
  k_copy_f32<<<blks(n64), TB, 0, stream>>>(feats, X[0], n64);

  // ---- ECA gating per branch ----
  k_zero_f32<<<blks(5 * 256), TB, 0, stream>>>(sums, 5 * 256);
  k_zero_f32<<<1, 64, 0, stream>>>(counts, 4);
  k_counts<<<blks(N), TB, 0, stream>>>(bidx, counts, N);
  for (int i = 0; i < 5; ++i)
    k_sums<<<blks(n64), TB, 0, stream>>>(X[i], bidx, sums + i * 256, n64);
  for (int i = 0; i < 5; ++i)
    k_gate<<<1, 256, 0, stream>>>(sums + i * 256, counts, F(35 + i), gates + i * 256);
  for (int i = 0; i < 5; ++i)
    k_apply<<<blks(n64), TB, 0, stream>>>(X[i], XB[i], gates + i * 256, bidx, n64);

  // ---- trans layers: wsum = sum_i conv1(relu(conv1(xg[i], t_i0)), t_i1) ----
  for (int i = 0; i < 5; ++i) {
    gemm(XB[i], nullptr, 1, Ptr[i * 2],     F(40 + i * 6 + 1), F(40 + i * 6 + 0),
         64, nullptr, T1, 1, 0);
    gemm(T1,    nullptr, 1, Ptr[i * 2 + 1], F(40 + i * 6 + 4), F(40 + i * 6 + 3),
         64, wsum, nullptr, 0, (i > 0) ? 1 : 0);
  }
  // w320 = conv1(wsum, trans2)  [N,320]
  k_f2bf<<<blks(n64), TB, 0, stream>>>(wsum, T1, n64);
  gemm(T1, nullptr, 1, Ptr2, F(71), F(70), 320, w320, nullptr, 0, 0);

  // ---- softmax over branches, top-k mask, gated combine ----
  k_final<<<blks(n64), TB, 0, stream>>>(w320, X[0], X[1], X[2], X[3], X[4],
                                        keep, (float*)d_out, n64);
  (void)n_in; (void)out_size; (void)ws_size;
}
